// EnergyPredictor_2937757631009
// MI455X (gfx1250) — compile-verified, matrix operand order checked by reading, not run on hardware
//
#include <hip/hip_runtime.h>
#include <hip/hip_bf16.h>
#include <math.h>

typedef __attribute__((ext_vector_type(16))) __bf16 v16bf;
typedef __attribute__((ext_vector_type(8)))  float  v8f;

#define N_NODES 50000
#define N_EDGES 800000
#define LDX 160            // padded leading dim (floats) for node buffers
#define TILE_BF 512        // 32 lanes * 16 bf16 per packed WMMA tile

__device__ __forceinline__ float silu_f(float v) { return v / (1.0f + __expf(-v)); }

__device__ __forceinline__ v8f wmma_bf16(v16bf a, v16bf b, v8f c) {
  return __builtin_amdgcn_wmma_f32_16x16x32_bf16(false, a, false, b, (short)0, c, false, false);
}

__device__ __forceinline__ v16bf cvt16(float4 r0, float4 r1, float4 r2, float4 r3) {
  v16bf a;
  a[0]  = (__bf16)r0.x; a[1]  = (__bf16)r0.y; a[2]  = (__bf16)r0.z; a[3]  = (__bf16)r0.w;
  a[4]  = (__bf16)r1.x; a[5]  = (__bf16)r1.y; a[6]  = (__bf16)r1.z; a[7]  = (__bf16)r1.w;
  a[8]  = (__bf16)r2.x; a[9]  = (__bf16)r2.y; a[10] = (__bf16)r2.z; a[11] = (__bf16)r2.w;
  a[12] = (__bf16)r3.x; a[13] = (__bf16)r3.y; a[14] = (__bf16)r3.z; a[15] = (__bf16)r3.w;
  return a;
}

// Guard-free A-tile load from a zero-padded row-major f32 matrix (ld mult of 32).
// ISA 7.12.2 layout: lane holds row M=lane&15; kbase=(lane<16?0:8);
// a[i] -> K = k0 + kbase + (i/8)*16 + (i%8)  => two runs of 8 consecutive floats.
__device__ __forceinline__ v16bf load_a_f32(const float* M, int ld, int row0, int k0) {
  int lane = threadIdx.x & 31;
  const float* p = M + (size_t)(row0 + (lane & 15)) * (size_t)ld + k0 + ((lane >> 4) << 3);
  float4 r0 = ((const float4*)p)[0];
  float4 r1 = ((const float4*)p)[1];
  float4 r2 = ((const float4*)(p + 16))[0];
  float4 r3 = ((const float4*)(p + 16))[1];
  return cvt16(r0, r1, r2, r3);
}

// Same, from LDS (ld exact multiple of 32, no padding needed)
__device__ __forceinline__ v16bf load_a_lds(const float* L, int ld, int k0) {
  int lane = threadIdx.x & 31;
  const float* p = L + (lane & 15) * ld + k0 + ((lane >> 4) << 3);
  float4 r0 = ((const float4*)p)[0];
  float4 r1 = ((const float4*)p)[1];
  float4 r2 = ((const float4*)(p + 16))[0];
  float4 r3 = ((const float4*)(p + 16))[1];
  return cvt16(r0, r1, r2, r3);
}

// Packed-tile load: tile = 32 lanes * 16 bf16 contiguous, one 32B load per lane.
__device__ __forceinline__ v16bf load_tile(const __bf16* P, int tile) {
  int lane = threadIdx.x & 31;
  return ((const v16bf*)P)[(size_t)tile * 32 + lane];
}

// ============================ pack kernels (run once per launch) ============================

// Pack B-layout tiles from row-major f32 weight [K x Nout], padding K->32*ksteps, N->16*ntiles.
__global__ void pack_b_kernel(const float* __restrict__ W, int ldn, int kmax, int nmax,
                              int ntiles, int total, __bf16* __restrict__ dst) {
  int idx = blockIdx.x * blockDim.x + threadIdx.x;
  if (idx >= total) return;
  int i    = idx & 15;
  int lane = (idx >> 4) & 31;
  int tile = idx >> 9;                 // kk*ntiles + nt
  int kk   = tile / ntiles;
  int nt   = tile - kk * ntiles;
  int n = nt * 16 + (lane & 15);
  int k = kk * 32 + ((lane >> 4) << 3) + ((i >> 3) << 4) + (i & 7);
  float v = (k < kmax && n < nmax) ? W[(size_t)k * ldn + n] : 0.0f;
  dst[idx] = (__bf16)v;
}

// Pack A-layout tiles from row-major f32 [rows x ld] (single K-step, K padded to 32).
__global__ void pack_a_kernel(const float* __restrict__ M, int ld, int kmax,
                              int total, __bf16* __restrict__ dst) {
  int idx = blockIdx.x * blockDim.x + threadIdx.x;
  if (idx >= total) return;
  int i    = idx & 15;
  int lane = (idx >> 4) & 31;
  int t    = idx >> 9;
  int row = t * 16 + (lane & 15);
  int k = ((lane >> 4) << 3) + ((i >> 3) << 4) + (i & 7);
  float v = (k < kmax) ? M[(size_t)row * ld + k] : 0.0f;
  dst[idx] = (__bf16)v;
}

// ============================ main kernels ============================

__global__ void init_x_kernel(const float* __restrict__ embed, float* __restrict__ x) {
  size_t idx = (size_t)blockIdx.x * blockDim.x + threadIdx.x;
  size_t total = (size_t)N_NODES * LDX;
  if (idx >= total) return;
  int c = (int)(idx % LDX);
  size_t n = idx / LDX;
  x[idx] = (c < 64) ? embed[n * 64 + c] : 0.0f;
}

__global__ void fill_zero_kernel(float* __restrict__ p, size_t total) {
  size_t idx = (size_t)blockIdx.x * blockDim.x + threadIdx.x;
  if (idx < total) p[idx] = 0.0f;
}

// One wave per block, one 16x16 output tile:
//   y = x @ W_lin ; s = x @ W_self + node_attr @ W_attr
__global__ void node_gemm_kernel(const float* __restrict__ x, const __bf16* __restrict__ attrA,
                                 const __bf16* __restrict__ BlinP, const __bf16* __restrict__ BselfP,
                                 const __bf16* __restrict__ BattrP,
                                 float* __restrict__ y, float* __restrict__ s,
                                 int ksteps, int ntiles, int d_out) {
  int row0 = blockIdx.x * 16;
  int nt   = blockIdx.y;
  v8f cy = {};
  v8f cs = {};
  for (int kk = 0; kk < ksteps; ++kk) {
    v16bf a = load_a_f32(x, LDX, row0, kk * 32);
    cy = wmma_bf16(a, load_tile(BlinP,  kk * ntiles + nt), cy);
    cs = wmma_bf16(a, load_tile(BselfP, kk * ntiles + nt), cs);
  }
  {
    v16bf a = load_tile(attrA, blockIdx.x);
    cs = wmma_bf16(a, load_tile(BattrP, nt), cs);
  }
  int lane = threadIdx.x & 31;
  int n = nt * 16 + (lane & 15);
  int rbase = row0 + ((lane >> 4) << 3);
  if (n < d_out) {
#pragma unroll
    for (int j = 0; j < 8; ++j) {
      y[(size_t)(rbase + j) * LDX + n] = cy[j];
      s[(size_t)(rbase + j) * LDX + n] = cs[j];
    }
  }
}

// One wave per block, 16 edges:
//   radial = silu(silu(elen@R1)@R2)@R3 ; m = (y[src] + eattr@W_edge)*radial ; atomicAdd agg[dst]
__global__ void edge_kernel(const float* __restrict__ y,
                            const __bf16* __restrict__ elenA, const __bf16* __restrict__ eattrA,
                            const int* __restrict__ esrc, const int* __restrict__ edst,
                            const __bf16* __restrict__ R1P, const __bf16* __restrict__ R2P,
                            const __bf16* __restrict__ R3P, const __bf16* __restrict__ WedgeP,
                            float* __restrict__ agg, int ntiles, int d_out) {
  __shared__ __align__(16) float h1[16 * 32];
  __shared__ __align__(16) float h2[16 * 64];
  __shared__ int s_src[16];
  __shared__ int s_dst[16];

  int t = blockIdx.x;
  int e0 = t * 16;
  int lane = threadIdx.x & 31;
  if (lane < 16) {
    s_src[lane] = esrc[e0 + lane];
    s_dst[lane] = edst[e0 + lane];
  }

  int col16 = lane & 15;
  int rb = (lane >> 4) << 3;

  // h1 = silu(elen @ R1) : 16x32
  {
    v16bf a = load_tile(elenA, t);
#pragma unroll
    for (int nt = 0; nt < 2; ++nt) {
      v8f c = {};
      c = wmma_bf16(a, load_tile(R1P, nt), c);
      int col = nt * 16 + col16;
#pragma unroll
      for (int j = 0; j < 8; ++j) h1[(rb + j) * 32 + col] = silu_f(c[j]);
    }
  }
  __syncthreads();

  // h2 = silu(h1 @ R2) : 16x64
  {
    v16bf a = load_a_lds(h1, 32, 0);
#pragma unroll
    for (int nt = 0; nt < 4; ++nt) {
      v8f c = {};
      c = wmma_bf16(a, load_tile(R2P, nt), c);
      int col = nt * 16 + col16;
#pragma unroll
      for (int j = 0; j < 8; ++j) h2[(rb + j) * 64 + col] = silu_f(c[j]);
    }
  }
  __syncthreads();

  v16bf a2_0 = load_a_lds(h2, 64, 0);
  v16bf a2_1 = load_a_lds(h2, 64, 32);
  v16bf a_at = load_tile(eattrA, t);

  for (int nt = 0; nt < ntiles; ++nt) {
    v8f cr = {};
    cr = wmma_bf16(a2_0, load_tile(R3P, nt), cr);
    cr = wmma_bf16(a2_1, load_tile(R3P, ntiles + nt), cr);
    v8f cw = {};
    cw = wmma_bf16(a_at, load_tile(WedgeP, nt), cw);

    int n = nt * 16 + col16;
    if (n < d_out) {
#pragma unroll
      for (int j = 0; j < 8; ++j) {
        int ei = rb + j;
        int src = s_src[ei];
        int dst = s_dst[ei];
        float yv = y[(size_t)src * LDX + n];     // L2-resident gather
        float m = (yv + cw[j]) * cr[j];
        atomicAdd(&agg[(size_t)dst * LDX + n], m);  // L2-resident scatter
      }
    }
  }
}

// out = silu?(agg/4 + s)
__global__ void combine_kernel(const float* __restrict__ agg, const float* __restrict__ s,
                               float* __restrict__ out, int ldo, int d_out, int activate) {
  size_t idx = (size_t)blockIdx.x * blockDim.x + threadIdx.x;
  size_t total = (size_t)N_NODES * (size_t)d_out;
  if (idx >= total) return;
  size_t n = idx / (size_t)d_out;
  int c = (int)(idx - n * (size_t)d_out);
  float v = agg[n * LDX + c] * 0.25f + s[n * LDX + c];
  if (activate) v = silu_f(v);
  out[n * (size_t)ldo + c] = v;
}

__global__ void graph_reduce_kernel(const float* __restrict__ fec, const int* __restrict__ batch,
                                    float* __restrict__ g) {
  size_t idx = (size_t)blockIdx.x * blockDim.x + threadIdx.x;
  if (idx >= (size_t)N_NODES * 8) return;
  size_t n = idx >> 3;
  int c = (int)(idx & 7);
  atomicAdd(&g[(size_t)batch[n] * 8 + c], fec[idx]);
}

__global__ void gumbel_softmax_kernel(const float* __restrict__ g, float* __restrict__ probs) {
  int gi = threadIdx.x;
  if (gi >= 64) return;
  float vals[8];
  float mx = -1e30f;
#pragma unroll
  for (int c = 0; c < 8; ++c) {
    unsigned h = (unsigned)(gi * 8 + c) * 2654435761u + 42u;
    h ^= h >> 16; h *= 2246822519u; h ^= h >> 13;
    float u = fmaxf((float)(h & 0xFFFFFFu) * (1.0f / 16777216.0f), 1e-6f);
    float gum = -logf(-logf(u));
    float v = (g[gi * 8 + c] + gum) * 0.01f;
    vals[c] = v;
    mx = fmaxf(mx, v);
  }
  float sum = 0.0f;
#pragma unroll
  for (int c = 0; c < 8; ++c) { vals[c] = expf(vals[c] - mx); sum += vals[c]; }
  float inv = 1.0f / sum;
#pragma unroll
  for (int c = 0; c < 8; ++c) probs[gi * 8 + c] = vals[c] * inv;
}

// ============================ launch ============================

extern "C" void kernel_launch(void* const* d_in, const int* in_sizes, int n_in,
                              void* d_out, int out_size, void* d_ws, size_t ws_size,
                              hipStream_t stream) {
  const float* embed = (const float*)d_in[0];
  const float* nattr = (const float*)d_in[1];
  const int*   esrc  = (const int*)d_in[2];
  const int*   edst  = (const int*)d_in[3];
  const float* eattr = (const float*)d_in[4];
  const float* elen  = (const float*)d_in[5];
  const int*   batch = (const int*)d_in[6];

  const size_t NB = (size_t)N_NODES * LDX;
  char* w = (char*)d_ws;
  float* x    = (float*)w; w += NB * 4;
  float* y    = (float*)w; w += NB * 4;
  float* sbuf = (float*)w; w += NB * 4;
  float* agg  = (float*)w; w += NB * 4;
  __bf16* attrA  = (__bf16*)w; w += (size_t)(N_NODES / 16) * TILE_BF * 2;
  __bf16* elenA  = (__bf16*)w; w += (size_t)(N_EDGES / 16) * TILE_BF * 2;
  __bf16* eattrA = (__bf16*)w; w += (size_t)(N_EDGES / 16) * TILE_BF * 2;
  __bf16* wpack[5];
  for (int L = 0; L < 5; ++L) { wpack[L] = (__bf16*)w; w += 256 * 1024; }
  float* g = (float*)w;

  float* out = (float*)d_out;           // [512 probs][400000 atoms_fec]
  float* fec = out + 512;

  const int THR = 256;
  const int dins[5]  = {64, 144, 144, 144, 144};
  const int douts[5] = {144, 144, 144, 144, 8};

  // ---- one-time packing ----
  init_x_kernel<<<(unsigned)((NB + THR - 1) / THR), THR, 0, stream>>>(embed, x);
  {
    int totA = (N_NODES / 16) * TILE_BF;
    pack_a_kernel<<<(totA + THR - 1) / THR, THR, 0, stream>>>(nattr, 16, 16, totA, attrA);
    int totE = (N_EDGES / 16) * TILE_BF;
    pack_a_kernel<<<(totE + THR - 1) / THR, THR, 0, stream>>>(elen, 10, 10, totE, elenA);
    pack_a_kernel<<<(totE + THR - 1) / THR, THR, 0, stream>>>(eattr, 9, 9, totE, eattrA);
  }

  __bf16 *R1P[5], *R2P[5], *R3P[5], *WeP[5], *WlP[5], *WsP[5], *WaP[5];
  for (int L = 0; L < 5; ++L) {
    int kin = (dins[L] + 31) >> 5;
    int nt  = (douts[L] + 15) >> 4;
    const float* R1    = (const float*)d_in[7 + L * 7 + 0];
    const float* R2    = (const float*)d_in[7 + L * 7 + 1];
    const float* R3    = (const float*)d_in[7 + L * 7 + 2];
    const float* Wlin  = (const float*)d_in[7 + L * 7 + 3];
    const float* Wedge = (const float*)d_in[7 + L * 7 + 4];
    const float* Wself = (const float*)d_in[7 + L * 7 + 5];
    const float* Wattr = (const float*)d_in[7 + L * 7 + 6];

    __bf16* b = wpack[L];
    R1P[L] = b;                         b += 2 * TILE_BF;
    R2P[L] = b;                         b += 4 * TILE_BF;
    R3P[L] = b;                         b += 2 * nt * TILE_BF;
    WeP[L] = b;                         b += nt * TILE_BF;
    WlP[L] = b;                         b += kin * nt * TILE_BF;
    WsP[L] = b;                         b += kin * nt * TILE_BF;
    WaP[L] = b;

    int tot;
    tot = 1 * 2 * TILE_BF;
    pack_b_kernel<<<(tot + THR - 1) / THR, THR, 0, stream>>>(R1, 32, 10, 32, 2, tot, R1P[L]);
    tot = 1 * 4 * TILE_BF;
    pack_b_kernel<<<(tot + THR - 1) / THR, THR, 0, stream>>>(R2, 64, 32, 64, 4, tot, R2P[L]);
    tot = 2 * nt * TILE_BF;
    pack_b_kernel<<<(tot + THR - 1) / THR, THR, 0, stream>>>(R3, douts[L], 64, douts[L], nt, tot, R3P[L]);
    tot = 1 * nt * TILE_BF;
    pack_b_kernel<<<(tot + THR - 1) / THR, THR, 0, stream>>>(Wedge, douts[L], 9, douts[L], nt, tot, WeP[L]);
    tot = kin * nt * TILE_BF;
    pack_b_kernel<<<(tot + THR - 1) / THR, THR, 0, stream>>>(Wlin, douts[L], dins[L], douts[L], nt, tot, WlP[L]);
    pack_b_kernel<<<(tot + THR - 1) / THR, THR, 0, stream>>>(Wself, douts[L], dins[L], douts[L], nt, tot, WsP[L]);
    tot = 1 * nt * TILE_BF;
    pack_b_kernel<<<(tot + THR - 1) / THR, THR, 0, stream>>>(Wattr, douts[L], 16, douts[L], nt, tot, WaP[L]);
  }

  // ---- 5 message-passing layers ----
  for (int L = 0; L < 5; ++L) {
    int kin = (dins[L] + 31) >> 5;
    int nt  = (douts[L] + 15) >> 4;

    node_gemm_kernel<<<dim3(N_NODES / 16, nt), 32, 0, stream>>>(
        x, attrA, WlP[L], WsP[L], WaP[L], y, sbuf, kin, nt, douts[L]);

    fill_zero_kernel<<<(unsigned)((NB + THR - 1) / THR), THR, 0, stream>>>(agg, NB);

    edge_kernel<<<N_EDGES / 16, 32, 0, stream>>>(
        y, elenA, eattrA, esrc, edst, R1P[L], R2P[L], R3P[L], WeP[L], agg, nt, douts[L]);

    if (L < 4) {
      size_t tot = (size_t)N_NODES * douts[L];
      combine_kernel<<<(unsigned)((tot + THR - 1) / THR), THR, 0, stream>>>(
          agg, sbuf, x, LDX, douts[L], 1);
    } else {
      size_t tot = (size_t)N_NODES * 8;
      combine_kernel<<<(unsigned)((tot + THR - 1) / THR), THR, 0, stream>>>(
          agg, sbuf, fec, 8, 8, 0);
    }
  }

  fill_zero_kernel<<<2, THR, 0, stream>>>(g, 512);
  graph_reduce_kernel<<<(unsigned)(((size_t)N_NODES * 8 + THR - 1) / THR), THR, 0, stream>>>(
      fec, batch, g);
  gumbel_softmax_kernel<<<1, 64, 0, stream>>>(g, out);
}
